// MaskMultiHeadAttention_91233695302254
// MI455X (gfx1250) — matrix operand order repaired and checked by reading, not compiled, not run on hardware
//
#include <hip/hip_runtime.h>
#include <hip/hip_bf16.h>

typedef _Float16 h4  __attribute__((ext_vector_type(4)));
typedef _Float16 h8  __attribute__((ext_vector_type(8)));
typedef _Float16 h16 __attribute__((ext_vector_type(16)));
typedef float    f8  __attribute__((ext_vector_type(8)));

union H16 { h16 v; h8 h[2]; };
union F8  { f8 v; float f[8]; };

#define Bn   4
#define Sdim 1024
#define Ddim 1024
#define Hn   16
#define DH   64

#define WMMA_F16(a, b, c) \
  __builtin_amdgcn_wmma_f32_16x16x32_f16(false, (a), false, (b), (short)0, (c), false, false)

// A operand (16x32 f16): lane holds row m=lane&15; halves = K in two 8-runs at
// base=8*(lane>>4) and base+16.  p points at (row, kbase + 8*(lane>>4)).
__device__ __forceinline__ h16 load_a16(const _Float16* p) {
  H16 u;
  u.h[0] = *(const h8*)p;
  u.h[1] = *(const h8*)(p + 16);
  return u.v;
}
// B operand (32x16 f16): lane holds column n=lane&15; halves = K in two 8-runs at
// base=8*(lane>>4) and base+16.  p points at (colrow, kbase + 8*(lane>>4)).
__device__ __forceinline__ h16 load_b16(const _Float16* p) {
  H16 u;
  u.h[0] = *(const h8*)p;
  u.h[1] = *(const h8*)(p + 16);
  return u.v;
}

// ---------------------------------------------------------------- utilities
__global__ void zero_kernel(float* __restrict__ p, int n) {
  int i = blockIdx.x * blockDim.x + threadIdx.x;
  if (i < n) p[i] = 0.0f;
}

__global__ void cvt4_kernel(const float4* __restrict__ in, _Float16* __restrict__ out, int n4) {
  int i = blockIdx.x * blockDim.x + threadIdx.x;
  if (i < n4) {
    float4 x = in[i];
    h4 y = { (_Float16)x.x, (_Float16)x.y, (_Float16)x.z, (_Float16)x.w };
    *(h4*)(out + 4 * (size_t)i) = y;
  }
}

// ---------------------------------------------------- additive mask precompute
// add[b,m,n] = -100 if (mask[b,m,n]>=.5) + (mask_ver[b,n,m]>=.5) == 1 else 0
// LDS-tiled so the transposed mask_ver read stays coalesced.
__global__ __launch_bounds__(1024) void maskprep_kernel(
    const float* __restrict__ mask, const float* __restrict__ mask_ver,
    _Float16* __restrict__ madd) {
  __shared__ float vt[32][33];
  const int b  = blockIdx.z;
  const int m0 = blockIdx.y * 32;
  const int n0 = blockIdx.x * 32;
  // vt[n_local][m_local] <- mask_ver[b, n0+n_local, m0+m_local] (coalesced rows)
  vt[threadIdx.y][threadIdx.x] =
      mask_ver[((size_t)b * Sdim + n0 + threadIdx.y) * Sdim + m0 + threadIdx.x];
  __syncthreads();
  const int m = m0 + threadIdx.y;
  const int n = n0 + threadIdx.x;
  const int flags = (mask[((size_t)b * Sdim + m) * Sdim + n] >= 0.5f)
                  + (vt[threadIdx.x][threadIdx.y] >= 0.5f);
  madd[((size_t)b * Sdim + m) * Sdim + n] =
      (flags == 1) ? (_Float16)(-100.0f) : (_Float16)(0.0f);
}

// ----------------------------------------------------- projection GEMM (f16 out)
// C[m,n] = sum_k X[m,k] * W[n,k] + bias[n]   (x @ W.T + b)
// Wave tile: 32 rows x 64 cols (two row sub-tiles share every B fetch).
// vlayout 0: out[((b*H+h)*S + s)*DH + c]   (Qh / Kh)
// vlayout 1: out[((b*H+h)*DH + c)*S + s]   (V transposed)
__global__ __launch_bounds__(128) void gemm_proj_kernel(
    const _Float16* __restrict__ X, const _Float16* __restrict__ Wt,
    const float* __restrict__ bias, _Float16* __restrict__ out, int vlayout) {
  const int w    = threadIdx.x >> 5;
  const int lane = threadIdx.x & 31;
  const int half = lane >> 4;
  const int l15  = lane & 15;
  const int m0   = blockIdx.x * 128 + w * 32;
  const int n0   = blockIdx.y * 64;

  F8 acc[2][4];
#pragma unroll
  for (int r = 0; r < 2; ++r)
#pragma unroll
    for (int t = 0; t < 4; ++t)
#pragma unroll
      for (int j = 0; j < 8; ++j) acc[r][t].f[j] = 0.0f;

  const _Float16* arow0 = X + (size_t)(m0 + l15) * Ddim;
  const _Float16* arow1 = X + (size_t)(m0 + 16 + l15) * Ddim;
  for (int kk = 0; kk < Ddim; kk += 32) {
    h16 a0 = load_a16(arow0 + kk + half * 8);
    h16 a1 = load_a16(arow1 + kk + half * 8);
#pragma unroll
    for (int t = 0; t < 4; ++t) {
      h16 bm = load_b16(Wt + (size_t)(n0 + t * 16 + l15) * Ddim + kk + half * 8);
      acc[0][t].v = WMMA_F16(a0, bm, acc[0][t].v);
      acc[1][t].v = WMMA_F16(a1, bm, acc[1][t].v);
    }
  }

#pragma unroll
  for (int t = 0; t < 4; ++t) {
    const int n  = n0 + t * 16 + l15;
    const float bv = bias[n];
    const int hh = n >> 6;       // head
    const int c  = n & 63;       // channel within head
#pragma unroll
    for (int r = 0; r < 2; ++r)
#pragma unroll
      for (int i = 0; i < 8; ++i) {
        const int m    = m0 + r * 16 + i + half * 8;
        const int bidx = m >> 10;  // batch
        const int s    = m & 1023; // sequence pos
        size_t off = (vlayout == 0)
            ? ((((size_t)bidx * Hn + hh) * Sdim + s) * DH + c)
            : ((((size_t)bidx * Hn + hh) * DH + c) * Sdim + s);
        out[off] = (_Float16)(acc[r][t].f[i] + bv);
      }
  }
}

// -------------------------------------------------------- output GEMM (f32 out)
__global__ __launch_bounds__(128) void gemm_out_kernel(
    const _Float16* __restrict__ X, const _Float16* __restrict__ Wt,
    const float* __restrict__ bias, float* __restrict__ out) {
  const int w    = threadIdx.x >> 5;
  const int lane = threadIdx.x & 31;
  const int half = lane >> 4;
  const int l15  = lane & 15;
  const int m0   = blockIdx.x * 128 + w * 32;
  const int n0   = blockIdx.y * 64;

  F8 acc[2][4];
#pragma unroll
  for (int r = 0; r < 2; ++r)
#pragma unroll
    for (int t = 0; t < 4; ++t)
#pragma unroll
      for (int j = 0; j < 8; ++j) acc[r][t].f[j] = 0.0f;

  const _Float16* arow0 = X + (size_t)(m0 + l15) * Ddim;
  const _Float16* arow1 = X + (size_t)(m0 + 16 + l15) * Ddim;
  for (int kk = 0; kk < Ddim; kk += 32) {
    h16 a0 = load_a16(arow0 + kk + half * 8);
    h16 a1 = load_a16(arow1 + kk + half * 8);
#pragma unroll
    for (int t = 0; t < 4; ++t) {
      h16 bm = load_b16(Wt + (size_t)(n0 + t * 16 + l15) * Ddim + kk + half * 8);
      acc[0][t].v = WMMA_F16(a0, bm, acc[0][t].v);
      acc[1][t].v = WMMA_F16(a1, bm, acc[1][t].v);
    }
  }

#pragma unroll
  for (int t = 0; t < 4; ++t) {
    const int n    = n0 + t * 16 + l15;
    const float bv = bias[n];
#pragma unroll
    for (int r = 0; r < 2; ++r)
#pragma unroll
      for (int i = 0; i < 8; ++i) {
        const int m = m0 + r * 16 + i + half * 8;
        out[(size_t)m * Ddim + n] = acc[r][t].f[i] + bv;
      }
  }
}

// -------------------------------------------------- flash attention per (b,h)
// One wave handles 16 query rows; streams 64-key blocks with online softmax.
__global__ __launch_bounds__(128) void attn_kernel(
    const _Float16* __restrict__ Qh, const _Float16* __restrict__ Kh,
    const _Float16* __restrict__ Vt, const _Float16* __restrict__ madd,
    _Float16* __restrict__ Yws, float* __restrict__ weight) {
  __shared__ __align__(32) _Float16 plds[4][16 * 64];

  const int w    = threadIdx.x >> 5;
  const int lane = threadIdx.x & 31;
  const int half = lane >> 4;
  const int l15  = lane & 15;

  const int wtile = blockIdx.x * 4 + w;     // 0 .. B*H*(S/16)-1
  const int qt    = wtile & 63;             // S/16 = 64
  const int bh    = wtile >> 6;
  const int b     = bh >> 4;                // H = 16
  const int h     = bh & 15;
  const int m0    = qt * 16;

  const _Float16* Qp = Qh + (size_t)bh * Sdim * DH;
  const _Float16* Kp = Kh + (size_t)bh * Sdim * DH;
  const _Float16* Vp = Vt + (size_t)bh * DH * Sdim;
  _Float16* pl = plds[w];

  h16 qa[2];
  qa[0] = load_a16(Qp + (size_t)(m0 + l15) * DH + 0  + half * 8);
  qa[1] = load_a16(Qp + (size_t)(m0 + l15) * DH + 32 + half * 8);

  F8 yacc[4];
#pragma unroll
  for (int t = 0; t < 4; ++t)
#pragma unroll
    for (int j = 0; j < 8; ++j) yacc[t].f[j] = 0.0f;

  float Mrow[8], Lrow[8], Wrow[8];
#pragma unroll
  for (int i = 0; i < 8; ++i) { Mrow[i] = -3.0e38f; Lrow[i] = 0.0f; Wrow[i] = 0.0f; }

  const float scale = 0.124999999984375f;  // 1 / (sqrt(64) + 1e-9)

  for (int kb = 0; kb < Sdim / 64; ++kb) {
    const int n0 = kb * 64;

    // S = Q @ K^T  (16 x 64)
    F8 sacc[4];
#pragma unroll
    for (int t = 0; t < 4; ++t) {
      h16 b0 = load_b16(Kp + (size_t)(n0 + t * 16 + l15) * DH + 0  + half * 8);
      h16 b1 = load_b16(Kp + (size_t)(n0 + t * 16 + l15) * DH + 32 + half * 8);
      f8 c;
#pragma unroll
      for (int j = 0; j < 8; ++j) c[j] = 0.0f;
      c = WMMA_F16(qa[0], b0, c);
      c = WMMA_F16(qa[1], b1, c);
      sacc[t].v = c;
    }

    // scale + precomputed additive mask + weight accumulation + online softmax
#pragma unroll
    for (int i = 0; i < 8; ++i) {
      const int m = m0 + i + half * 8;
      const _Float16* arow = madd + ((size_t)b * Sdim + m) * Sdim;
      float rm = Mrow[i];
#pragma unroll
      for (int t = 0; t < 4; ++t) {
        const int n = n0 + t * 16 + l15;
        float s = sacc[t].f[i] * scale + (float)arow[n];
        sacc[t].f[i] = s;
        Wrow[i] += s;                       // masked pre-softmax score sum
        rm = fmaxf(rm, s);
      }
#pragma unroll
      for (int d = 1; d < 16; d <<= 1) rm = fmaxf(rm, __shfl_xor(rm, d, 32));
      const float alpha = __expf(Mrow[i] - rm);
      Mrow[i] = rm;
      Lrow[i] *= alpha;
#pragma unroll
      for (int t = 0; t < 4; ++t) yacc[t].f[i] *= alpha;
      float psum = 0.0f;
#pragma unroll
      for (int t = 0; t < 4; ++t) {
        const float p = __expf(sacc[t].f[i] - rm);
        psum += p;
        pl[(i + half * 8) * 64 + t * 16 + l15] = (_Float16)p;   // C-layout -> LDS
      }
#pragma unroll
      for (int d = 1; d < 16; d <<= 1) psum += __shfl_xor(psum, d, 32);
      Lrow[i] += psum;
    }

    // Y += P @ V  (read P back from LDS in A-operand layout)
#pragma unroll
    for (int ks = 0; ks < 2; ++ks) {
      H16 pa;
      pa.h[0] = *(const h8*)(pl + l15 * 64 + ks * 32 + half * 8);
      pa.h[1] = *(const h8*)(pl + l15 * 64 + ks * 32 + half * 8 + 16);
#pragma unroll
      for (int t = 0; t < 4; ++t) {
        h16 vb = load_b16(Vp + (size_t)(t * 16 + l15) * Sdim + n0 + ks * 32 + half * 8);
        yacc[t].v = WMMA_F16(pa.v, vb, yacc[t].v);
      }
    }
  }

  // epilogue: normalize, store Y (f16, merged-head layout), reduce weight rows
#pragma unroll
  for (int i = 0; i < 8; ++i) {
    const int srow = m0 + i + half * 8;
    const float inv = 1.0f / Lrow[i];
#pragma unroll
    for (int t = 0; t < 4; ++t) {
      Yws[((size_t)b * Sdim + srow) * Ddim + h * 64 + t * 16 + l15] =
          (_Float16)(yacc[t].f[i] * inv);
    }
    float wsum = Wrow[i];
#pragma unroll
    for (int d = 1; d < 16; d <<= 1) wsum += __shfl_xor(wsum, d, 32);
    if (l15 == 0)
      atomicAdd(weight + (size_t)b * Sdim + srow, wsum * (1.0f / (Hn * Sdim)));
  }
}

// ------------------------------------------------------------------ launcher
extern "C" void kernel_launch(void* const* d_in, const int* in_sizes, int n_in,
                              void* d_out, int out_size, void* d_ws, size_t ws_size,
                              hipStream_t stream) {
  (void)in_sizes; (void)n_in; (void)out_size; (void)ws_size;
  const float* q        = (const float*)d_in[0];
  const float* k        = (const float*)d_in[1];
  const float* v        = (const float*)d_in[2];
  const float* mask     = (const float*)d_in[3];
  const float* mask_ver = (const float*)d_in[4];
  const float* Wq = (const float*)d_in[5];
  const float* bq = (const float*)d_in[6];
  const float* Wk = (const float*)d_in[7];
  const float* bk = (const float*)d_in[8];
  const float* Wv = (const float*)d_in[9];
  const float* bv = (const float*)d_in[10];
  const float* Wo = (const float*)d_in[11];
  const float* bo = (const float*)d_in[12];

  const size_t BSD = (size_t)Bn * Sdim * Ddim;  // 4 Mi elements
  const size_t DD  = (size_t)Ddim * Ddim;       // 1 Mi elements
  const size_t BSS = (size_t)Bn * Sdim * Sdim;  // 4 Mi elements

  _Float16* ws16 = (_Float16*)d_ws;
  _Float16* q16  = ws16;
  _Float16* k16  = q16 + BSD;
  _Float16* v16  = k16 + BSD;
  _Float16* Wq16 = v16 + BSD;
  _Float16* Wk16 = Wq16 + DD;
  _Float16* Wv16 = Wk16 + DD;
  _Float16* Wo16 = Wv16 + DD;
  _Float16* Qhh  = Wo16 + DD;   // [B,H,S,DH] f16
  _Float16* Khh  = Qhh + BSD;   // [B,H,S,DH] f16
  _Float16* Vtt  = Khh + BSD;   // [B,H,DH,S] f16
  _Float16* Yws  = Vtt + BSD;   // [B,S,D]   f16 (pre-Wo)
  _Float16* Madd = Yws + BSD;   // [B,S,S]   f16 additive mask

  float* out    = (float*)d_out;
  float* weight = out + BSD;    // [B,S] second output

  zero_kernel<<<dim3((Bn * Sdim + 255) / 256), dim3(256), 0, stream>>>(weight, Bn * Sdim);

  const int nBSD4 = (int)(BSD / 4), nDD4 = (int)(DD / 4);
  cvt4_kernel<<<dim3((nBSD4 + 255) / 256), dim3(256), 0, stream>>>((const float4*)q, q16, nBSD4);
  cvt4_kernel<<<dim3((nBSD4 + 255) / 256), dim3(256), 0, stream>>>((const float4*)k, k16, nBSD4);
  cvt4_kernel<<<dim3((nBSD4 + 255) / 256), dim3(256), 0, stream>>>((const float4*)v, v16, nBSD4);
  cvt4_kernel<<<dim3((nDD4 + 255) / 256), dim3(256), 0, stream>>>((const float4*)Wq, Wq16, nDD4);
  cvt4_kernel<<<dim3((nDD4 + 255) / 256), dim3(256), 0, stream>>>((const float4*)Wk, Wk16, nDD4);
  cvt4_kernel<<<dim3((nDD4 + 255) / 256), dim3(256), 0, stream>>>((const float4*)Wv, Wv16, nDD4);
  cvt4_kernel<<<dim3((nDD4 + 255) / 256), dim3(256), 0, stream>>>((const float4*)Wo, Wo16, nDD4);

  maskprep_kernel<<<dim3(Sdim / 32, Sdim / 32, Bn), dim3(32, 32), 0, stream>>>(
      mask, mask_ver, Madd);

  dim3 gg(32, 16), bb(128);   // M=4096 (32 blocks x 4 waves x 32 rows), N=1024
  gemm_proj_kernel<<<gg, bb, 0, stream>>>(q16, Wq16, bq, Qhh, 0);
  gemm_proj_kernel<<<gg, bb, 0, stream>>>(k16, Wk16, bk, Khh, 0);
  gemm_proj_kernel<<<gg, bb, 0, stream>>>(v16, Wv16, bv, Vtt, 1);

  attn_kernel<<<dim3(Bn * Hn * (Sdim / 16) / 4), bb, 0, stream>>>(
      Qhh, Khh, Vtt, Madd, Yws, weight);

  gemm_out_kernel<<<gg, bb, 0, stream>>>(Yws, Wo16, bo, out);
}
